// Relative_multi_head_attention_34565896798695
// MI455X (gfx1250) — matrix operand order repaired and checked by reading, not compiled, not run on hardware
//
#include <hip/hip_runtime.h>
#include <hip/hip_bf16.h>

#define BATCH  4
#define SEQL   1024
#define FULLL  2048
#define DMODEL 512
#define NHEADS 8
#define DHEAD  64

typedef __attribute__((ext_vector_type(16))) __bf16 v16bf;
typedef __attribute__((ext_vector_type(8)))  float  v8f;

// ---------------------------------------------------------------------------
// Transpose + f32->bf16 convert (for weight matrices: W[K,N] -> Wt[N,K] bf16)
// ---------------------------------------------------------------------------
__global__ void __launch_bounds__(256) transpose_cvt_bf16(
    const float* __restrict__ W, __bf16* __restrict__ Wt, int R, int C) {
  __shared__ float tile[32][33];
  int c0 = blockIdx.x * 32, r0 = blockIdx.y * 32;
  int tx = threadIdx.x & 31, ty = threadIdx.x >> 5;
  #pragma unroll
  for (int i = ty; i < 32; i += 8)
    tile[i][tx] = W[(size_t)(r0 + i) * C + (c0 + tx)];
  __syncthreads();
  #pragma unroll
  for (int i = ty; i < 32; i += 8)
    Wt[(size_t)(c0 + i) * R + (r0 + tx)] = (__bf16)tile[tx][i];
}

// ---------------------------------------------------------------------------
// Operand helpers for the WMMA GEMM
// ---------------------------------------------------------------------------
__device__ __forceinline__ v16bf cvtA(const float4* s) {
  v16bf av;
  av[0]=(__bf16)s[0].x; av[1]=(__bf16)s[0].y; av[2] =(__bf16)s[0].z; av[3] =(__bf16)s[0].w;
  av[4]=(__bf16)s[1].x; av[5]=(__bf16)s[1].y; av[6] =(__bf16)s[1].z; av[7] =(__bf16)s[1].w;
  av[8]=(__bf16)s[2].x; av[9]=(__bf16)s[2].y; av[10]=(__bf16)s[2].z; av[11]=(__bf16)s[2].w;
  av[12]=(__bf16)s[3].x; av[13]=(__bf16)s[3].y; av[14]=(__bf16)s[3].z; av[15]=(__bf16)s[3].w;
  return av;
}
// A(16x32 bf16) lane layout: K = e + 16*h -> one contiguous 16-float run
__device__ __forceinline__ void loadA(float4* dst, const float* aPtr, int kk, int h) {
  dst[0] = *(const float4*)(aPtr + kk + 16 * h);
  dst[1] = *(const float4*)(aPtr + kk + 16 * h + 4);
  dst[2] = *(const float4*)(aPtr + kk + 16 * h + 8);
  dst[3] = *(const float4*)(aPtr + kk + 16 * h + 12);
}
// B(32x16 bf16) lane layout: col = l16, K = e + 16*h -> one 32B load per subtile
__device__ __forceinline__ void loadB(v16bf* dst, const __bf16* bPtr, int kk, size_t step) {
  dst[0] = *(const v16bf*)(bPtr + kk);
  dst[1] = *(const v16bf*)(bPtr + kk + step);
  dst[2] = *(const v16bf*)(bPtr + kk + 2 * step);
  dst[3] = *(const v16bf*)(bPtr + kk + 3 * step);
}
#define WMMA4(av, bset)                                                                   \
  acc0 = __builtin_amdgcn_wmma_f32_16x16x32_bf16(false, av, false, bset[0], (short)0, acc0, false, false); \
  acc1 = __builtin_amdgcn_wmma_f32_16x16x32_bf16(false, av, false, bset[1], (short)0, acc1, false, false); \
  acc2 = __builtin_amdgcn_wmma_f32_16x16x32_bf16(false, av, false, bset[2], (short)0, acc2, false, false); \
  acc3 = __builtin_amdgcn_wmma_f32_16x16x32_bf16(false, av, false, bset[3], (short)0, acc3, false, false);

// ---------------------------------------------------------------------------
// Generic WMMA GEMM:  C[M,N] = A[M,K](f32, cvt->bf16 inline) @ Bt[N,K](bf16)^T
// Block: 256 threads = 8 waves, 4(M) x 2(N). Wave tile 16x64: 4 unconditional
// WMMAs sharing one A operand. K-loop software-pipelined (unroll 2, double-
// buffered operand registers, peeled tail) so next-step loads issue before the
// WMMAs consuming the current step. K must be a multiple of 64 (64/512/2048).
// Batched over gridDim.z: zb = z >> nhShift, zn = z & ((1<<nhShift)-1).
// A row remap (for inputs[:, -SEQ:, :]): phys = row*lda + (row>>rowShift)*rowGap
// MODE: 0=f32  1=f32+bias  2=bf16  3=bf16 v-transposed [b,h,d,f]  4=f32 gelu
// ---------------------------------------------------------------------------
template <int MODE>
__global__ void __launch_bounds__(256) gemm_f32a_bf16bt(
    const float* __restrict__ A, const __bf16* __restrict__ Bt,
    void* __restrict__ Cout, const float* __restrict__ bias,
    int M, int N, int K, int lda, int ldbt, int ldc,
    long long aBatB, long long aBatN,
    long long bBatB, long long bBatN,
    long long cBatB, long long cBatN,
    int nhShift, int rowShift, long long rowGap)
{
  int z  = blockIdx.z;
  int zb = z >> nhShift, zn = z & ((1 << nhShift) - 1);
  const float*  Ab = A  + (size_t)zb * aBatB + (size_t)zn * aBatN;
  const __bf16* Bb = Bt + (size_t)zb * bBatB + (size_t)zn * bBatN;

  int lane = threadIdx.x & 31;
  int wv   = threadIdx.x >> 5;
  int h    = lane >> 4;          // half-wave select
  int l16  = lane & 15;
  int m0   = blockIdx.y * 64  + (wv & 3) * 16;
  int n0   = blockIdx.x * 128 + (wv >> 2) * 64;
  if (m0 >= M || n0 >= N) return;   // wave-uniform

  int rowA = m0 + l16;           // A operand: lane l16 holds row m0+l16
  const float* aPtr = Ab + (size_t)rowA * lda + (size_t)(rowA >> rowShift) * rowGap;
  const __bf16* bPtr = Bb + (size_t)(n0 + l16) * ldbt + 16 * h;
  size_t bStep = (size_t)16 * ldbt;   // step between 16-col subtiles

  v8f acc0, acc1, acc2, acc3;
  #pragma unroll
  for (int i = 0; i < 8; ++i) { acc0[i] = 0.f; acc1[i] = 0.f; acc2[i] = 0.f; acc3[i] = 0.f; }

  float4 a0[4], a1[4];
  v16bf  b0[4], b1[4];
  loadA(a0, aPtr, 0, h);
  loadB(b0, bPtr, 0, bStep);

  int k0 = 0;
  for (; k0 + 64 < K; k0 += 64) {
    loadA(a1, aPtr, k0 + 32, h);
    loadB(b1, bPtr, k0 + 32, bStep);
    { v16bf av = cvtA(a0); WMMA4(av, b0); }
    loadA(a0, aPtr, k0 + 64, h);
    loadB(b0, bPtr, k0 + 64, bStep);
    { v16bf av = cvtA(a1); WMMA4(av, b1); }
  }
  // tail: k0 + 64 == K
  loadA(a1, aPtr, k0 + 32, h);
  loadB(b1, bPtr, k0 + 32, bStep);
  { v16bf av = cvtA(a0); WMMA4(av, b0); }
  { v16bf av = cvtA(a1); WMMA4(av, b1); }

  size_t cOffZ = (size_t)zb * cBatB + (size_t)zn * cBatN;
  #pragma unroll
  for (int t = 0; t < 4; ++t) {
    v8f acc = (t == 0) ? acc0 : (t == 1) ? acc1 : (t == 2) ? acc2 : acc3;
    #pragma unroll
    for (int i = 0; i < 8; ++i) {
      int row = m0 + 8 * h + i;            // C/D layout: M = i + 8*h
      int col = n0 + t * 16 + l16;         // N = l16
      float val = acc[i];
      if (MODE == 0) {
        ((float*)Cout)[cOffZ + (size_t)row * ldc + col] = val;
      } else if (MODE == 1) {
        ((float*)Cout)[cOffZ + (size_t)row * ldc + col] = val + bias[col];
      } else if (MODE == 2) {
        ((__bf16*)Cout)[cOffZ + (size_t)row * ldc + col] = (__bf16)val;
      } else if (MODE == 3) {              // v^T store: [b, h, d, f]
        int bb = row >> 11, f = row & (FULLL - 1);
        int nh = col >> 6,  d = col & (DHEAD - 1);
        size_t idx = ((size_t)((bb * NHEADS + nh) * DHEAD + d)) * FULLL + f;
        ((__bf16*)Cout)[idx] = (__bf16)val;
      } else {                             // exact GELU
        float g = 0.5f * val * (1.0f + erff(val * 0.70710678118654752f));
        ((float*)Cout)[cOffZ + (size_t)row * ldc + col] = g;
      }
    }
  }
}

// ---------------------------------------------------------------------------
// Fused rel-shift + mask + softmax + per-(b,h) attn max. One block per row.
// attn row holds A_C on entry; B_D[s,f] = P[s, f + (SEQ-1) - s] (valid where
// f <= s + mem_len; elsewhere masked -> attn = 0, matching f32 underflow).
// ---------------------------------------------------------------------------
__global__ void __launch_bounds__(256) softmax_rel(
    float* __restrict__ attn, const float* __restrict__ P, int* __restrict__ bnmax)
{
  __shared__ float sm[8];
  int s = blockIdx.x, n = blockIdx.y, b = blockIdx.z;
  size_t rowOff = ((size_t)((b * NHEADS + n) * SEQL + s)) * FULLL;
  float* rowA = attn + rowOff;
  const float* rowP = P + rowOff;
  const float scale = 0.04419417382415922f;  // 1/sqrt(512)
  int limit = s + (FULLL - SEQL);            // mem_len = 1024
  int shift = SEQL - 1 - s;

  float v[8];
  float mx = -3.0e38f;
  #pragma unroll
  for (int j = 0; j < 8; ++j) {
    int f = threadIdx.x + j * 256;
    if (f <= limit) {
      float x = (rowA[f] + rowP[f + shift]) * scale;
      v[j] = x; mx = fmaxf(mx, x);
    } else v[j] = -3.0e38f;
  }
  for (int o = 16; o; o >>= 1) mx = fmaxf(mx, __shfl_xor(mx, o, 32));
  if ((threadIdx.x & 31) == 0) sm[threadIdx.x >> 5] = mx;
  __syncthreads();
  float bm = fmaxf(fmaxf(fmaxf(sm[0], sm[1]), fmaxf(sm[2], sm[3])),
                   fmaxf(fmaxf(sm[4], sm[5]), fmaxf(sm[6], sm[7])));
  __syncthreads();

  float sum = 0.f;
  #pragma unroll
  for (int j = 0; j < 8; ++j) {
    int f = threadIdx.x + j * 256;
    float e = (f <= limit) ? __expf(v[j] - bm) : 0.f;
    v[j] = e; sum += e;
  }
  for (int o = 16; o; o >>= 1) sum += __shfl_xor(sum, o, 32);
  if ((threadIdx.x & 31) == 0) sm[threadIdx.x >> 5] = sum;
  __syncthreads();
  float bs = sm[0]+sm[1]+sm[2]+sm[3]+sm[4]+sm[5]+sm[6]+sm[7];
  float inv = 1.0f / bs;                     // row max of attn == exp(0)/sum == inv
  #pragma unroll
  for (int j = 0; j < 8; ++j)
    rowA[threadIdx.x + j * 256] = v[j] * inv;
  if (threadIdx.x == 0)
    atomicMax(bnmax + (b * NHEADS + n), __float_as_int(inv)); // positive floats
}

__global__ void init32(int* p) { p[threadIdx.x] = 0; }

__global__ void loss_mean(const int* __restrict__ bnmax, float* __restrict__ out) {
  float v = __int_as_float(bnmax[threadIdx.x]);
  for (int o = 16; o; o >>= 1) v += __shfl_xor(v, o, 32);
  if (threadIdx.x == 0) *out = v * (1.0f / 32.0f);
}

// ---------------------------------------------------------------------------
extern "C" void kernel_launch(void* const* d_in, const int* in_sizes, int n_in,
                              void* d_out, int out_size, void* d_ws, size_t ws_size,
                              hipStream_t stream) {
  const float* inputs  = (const float*)d_in[0];
  const float* rel_enc = (const float*)d_in[3];
  const float* Wq  = (const float*)d_in[4];
  const float* Wke = (const float*)d_in[5];
  const float* Wkr = (const float*)d_in[6];
  const float* Wv  = (const float*)d_in[7];
  const float* Wf  = (const float*)d_in[8];
  const float* u_p = (const float*)d_in[9];   // 512 = exactly a bias over N
  const float* v_p = (const float*)d_in[10];

  unsigned char* wsb = (unsigned char*)d_ws;
  size_t off = 0;
  auto take = [&](size_t bytes) -> void* {
    void* p = wsb + off; off += (bytes + 255) & ~(size_t)255; return p;
  };
  __bf16* Wqt  = (__bf16*)take((size_t)DMODEL * DMODEL * 2);
  __bf16* Wket = (__bf16*)take((size_t)DMODEL * DMODEL * 2);
  __bf16* Wkrt = (__bf16*)take((size_t)DMODEL * DMODEL * 2);
  __bf16* Wvt  = (__bf16*)take((size_t)DMODEL * DMODEL * 2);
  __bf16* Wft  = (__bf16*)take((size_t)DMODEL * DMODEL * 2);
  float*  qu   = (float*) take((size_t)BATCH * SEQL * DMODEL * 4);
  float*  qv   = (float*) take((size_t)BATCH * SEQL * DMODEL * 4);
  __bf16* Kbf  = (__bf16*)take((size_t)BATCH * FULLL * DMODEL * 2);
  __bf16* Vt   = (__bf16*)take((size_t)BATCH * FULLL * DMODEL * 2);
  __bf16* Qrb  = (__bf16*)take((size_t)FULLL * DMODEL * 2);
  float*  outf = (float*) take((size_t)BATCH * SEQL * DMODEL * 4);
  int*    bnmx = (int*)   take(32 * 4);
  float*  P    = (float*) take((size_t)BATCH * NHEADS * SEQL * FULLL * 4);
  (void)ws_size; (void)in_sizes; (void)n_in;

  float* outP  = (float*)d_out;
  float* attnP = outP + (size_t)BATCH * SEQL * DMODEL;
  float* lossP = outP + (size_t)out_size - 1;

  dim3 blk(256);
  dim3 tg(DMODEL / 32, DMODEL / 32);
  transpose_cvt_bf16<<<tg, blk, 0, stream>>>(Wq,  Wqt,  DMODEL, DMODEL);
  transpose_cvt_bf16<<<tg, blk, 0, stream>>>(Wke, Wket, DMODEL, DMODEL);
  transpose_cvt_bf16<<<tg, blk, 0, stream>>>(Wkr, Wkrt, DMODEL, DMODEL);
  transpose_cvt_bf16<<<tg, blk, 0, stream>>>(Wv,  Wvt,  DMODEL, DMODEL);
  transpose_cvt_bf16<<<tg, blk, 0, stream>>>(Wf,  Wft,  DMODEL, DMODEL);
  init32<<<1, 32, 0, stream>>>(bnmx);

  auto G = [](int M, int N, int Z) { return dim3((N + 127) / 128, (M + 63) / 64, Z); };
  const int NOSPLIT = 30;   // rowShift for "no remap" (row>>30 == 0)

  // qu = x @ Wq + u ; qv = x @ Wq + v   (x = inputs[:, -SEQ:, :] via row remap)
  gemm_f32a_bf16bt<1><<<G(BATCH*SEQL, DMODEL, 1), blk, 0, stream>>>(
      inputs + (size_t)SEQL * DMODEL, Wqt, qu, u_p,
      BATCH*SEQL, DMODEL, DMODEL, DMODEL, DMODEL, DMODEL,
      0,0,0,0,0,0, 0, 10, (long long)SEQL * DMODEL);
  gemm_f32a_bf16bt<1><<<G(BATCH*SEQL, DMODEL, 1), blk, 0, stream>>>(
      inputs + (size_t)SEQL * DMODEL, Wqt, qv, v_p,
      BATCH*SEQL, DMODEL, DMODEL, DMODEL, DMODEL, DMODEL,
      0,0,0,0,0,0, 0, 10, (long long)SEQL * DMODEL);
  // k (bf16, natural layout == Bt form for A_C) and v (bf16, transposed [b,h,d,f])
  gemm_f32a_bf16bt<2><<<G(BATCH*FULLL, DMODEL, 1), blk, 0, stream>>>(
      inputs, Wket, Kbf, nullptr,
      BATCH*FULLL, DMODEL, DMODEL, DMODEL, DMODEL, DMODEL,
      0,0,0,0,0,0, 0, NOSPLIT, 0);
  gemm_f32a_bf16bt<3><<<G(BATCH*FULLL, DMODEL, 1), blk, 0, stream>>>(
      inputs, Wvt, Vt, nullptr,
      BATCH*FULLL, DMODEL, DMODEL, DMODEL, DMODEL, DMODEL,
      0,0,0,0,0,0, 0, NOSPLIT, 0);
  // Qr = rel_enc @ Wkr (bf16 [r, h*64+d])
  gemm_f32a_bf16bt<2><<<G(FULLL, DMODEL, 1), blk, 0, stream>>>(
      rel_enc, Wkrt, Qrb, nullptr,
      FULLL, DMODEL, DMODEL, DMODEL, DMODEL, DMODEL,
      0,0,0,0,0,0, 0, NOSPLIT, 0);
  // A_C = (q+u) @ k^T  -> attn region of d_out (f32), batched over 32 (b,h)
  gemm_f32a_bf16bt<0><<<G(SEQL, FULLL, BATCH*NHEADS), blk, 0, stream>>>(
      qu, Kbf, attnP, nullptr,
      SEQL, FULLL, DHEAD, DMODEL, DMODEL, FULLL,
      (long long)SEQL * DMODEL, DHEAD,
      (long long)FULLL * DMODEL, DHEAD,
      (long long)NHEADS * SEQL * FULLL, (long long)SEQL * FULLL,
      3, NOSPLIT, 0);
  // P = (q+v) @ Qr^T -> ws (f32), Qr shared across batch
  gemm_f32a_bf16bt<0><<<G(SEQL, FULLL, BATCH*NHEADS), blk, 0, stream>>>(
      qv, Qrb, P, nullptr,
      SEQL, FULLL, DHEAD, DMODEL, DMODEL, FULLL,
      (long long)SEQL * DMODEL, DHEAD,
      0, DHEAD,
      (long long)NHEADS * SEQL * FULLL, (long long)SEQL * FULLL,
      3, NOSPLIT, 0);
  // fused rel-shift + mask + softmax + per-(b,h) max
  softmax_rel<<<dim3(SEQL, NHEADS, BATCH), blk, 0, stream>>>(attnP, P, bnmx);
  // out = attn @ v  (Bt = v^T[b,h,d,f])
  gemm_f32a_bf16bt<0><<<G(SEQL, DHEAD, BATCH*NHEADS), blk, 0, stream>>>(
      attnP, Vt, outf, nullptr,
      SEQL, DHEAD, FULLL, FULLL, FULLL, DMODEL,
      (long long)NHEADS * SEQL * FULLL, (long long)SEQL * FULLL,
      (long long)NHEADS * DHEAD * FULLL, (long long)DHEAD * FULLL,
      (long long)SEQL * DMODEL, DHEAD,
      3, NOSPLIT, 0);
  // output = gelu(out @ Wf)
  gemm_f32a_bf16bt<4><<<G(BATCH*SEQL, DMODEL, 1), blk, 0, stream>>>(
      outf, Wft, outP, nullptr,
      BATCH*SEQL, DMODEL, DMODEL, DMODEL, DMODEL, DMODEL,
      0,0,0,0,0,0, 0, NOSPLIT, 0);
  loss_mean<<<1, 32, 0, stream>>>(bnmx, lossP);
}
